// ButterflyLayer_40441412059167
// MI455X (gfx1250) — matrix operand order repaired and checked by reading, not compiled, not run on hardware
//
#include <hip/hip_runtime.h>

// ---------------------------------------------------------------------------
// Problem constants (from the reference)
// ---------------------------------------------------------------------------
#define BB      512          // batch
#define NPOS    1024         // positions (K*L == 1024 at every level)
#define CC      64           // channels
#define IN_F    8
#define OUT_F   8
#define TOTAL_BR 2046

typedef __attribute__((ext_vector_type(16))) __bf16        v16bf;
typedef __attribute__((ext_vector_type(8)))  float         v8f;
typedef __attribute__((ext_vector_type(4)))  unsigned int  v4u;

union FragBF { v16bf v; v4u u[2]; };

// Raw LDS byte offset of a __shared__ pointer: generic LDS address is
// {shared_aperture, lds_offset}; hardware truncates to addr[31:0].
static __device__ __forceinline__ unsigned lds_offset(const void* p)
{
    return (unsigned)(unsigned long long)p;
}

// 32(M) x 64(N) x 32*KT(K) bf16 GEMM core: two A tiles share each B fragment.
// A rows follow the 16-bit A 16x32 lane layout (low lanes K{0..7,16..23},
// high lanes K{8..15,24..31}); B fragments are pre-swizzled, lane-contiguous.
template<int KT>
static __device__ __forceinline__ void gemm32x64(
    const __bf16* arow0, const __bf16* arow1, const __bf16* wlds,
    int lane, v8f acc[2][4])
{
    const int half8 = (lane >> 4) << 3;
    #pragma unroll
    for (int kt = 0; kt < KT; ++kt) {
        FragBF a0, a1;
        a0.u[0] = *(const v4u*)(arow0 + kt * 32 + half8);
        a0.u[1] = *(const v4u*)(arow0 + kt * 32 + 16 + half8);
        a1.u[0] = *(const v4u*)(arow1 + kt * 32 + half8);
        a1.u[1] = *(const v4u*)(arow1 + kt * 32 + 16 + half8);
        #pragma unroll
        for (int nt = 0; nt < 4; ++nt) {
            FragBF bf;
            const v4u* wp = (const v4u*)(wlds + (kt * 4 + nt) * 512 + lane * 16);
            bf.u[0] = wp[0];
            bf.u[1] = wp[1];
            acc[0][nt] = __builtin_amdgcn_wmma_f32_16x16x32_bf16(
                false, a0.v, false, bf.v, (short)0, acc[0][nt], false, false);
            acc[1][nt] = __builtin_amdgcn_wmma_f32_16x16x32_bf16(
                false, a1.v, false, bf.v, (short)0, acc[1][nt], false, false);
        }
    }
}

static __device__ __forceinline__ void zero_acc(v8f acc[2][4])
{
    #pragma unroll
    for (int u = 0; u < 2; ++u)
        #pragma unroll
        for (int i = 0; i < 4; ++i)
            acc[u][i] = (v8f){0.f, 0.f, 0.f, 0.f, 0.f, 0.f, 0.f, 0.f};
}

// ---------------------------------------------------------------------------
// Weight pre-pack: filters (TOTAL_BR,2,64,64) f32 -> per-branch 128x64 bf16
// in WMMA B-fragment order:  Wp[((cc*4+kt)*4+nt)*512 + lane*16 + e]
// B 32x16 bf16 layout: lanes 0-15 hold K{0..7,16..23}, lanes 16-31
// hold K{8..15,24..31}; N = nt*16 + lane%16.  K-dim q = i*64 + c (pair elem, chan).
// ---------------------------------------------------------------------------
__global__ __launch_bounds__(256) void pack_filters_kernel(
    const float* __restrict__ filters, __bf16* __restrict__ Wp)
{
    unsigned idx  = blockIdx.x * 256u + threadIdx.x;   // < 2046*8192
    unsigned e    = idx & 15u;
    unsigned lane = (idx >> 4) & 31u;
    unsigned nt   = (idx >> 9) & 3u;
    unsigned kt   = (idx >> 11) & 3u;
    unsigned cc   = idx >> 13;
    unsigned q    = kt * 32u + ((e >> 3) << 4) + ((lane >> 4) << 3) + (e & 7u);  // 0..127
    unsigned i    = q >> 6;
    unsigned c    = q & 63u;
    unsigned n    = nt * 16u + (lane & 15u);
    float w = filters[(((size_t)cc * 2u + i) * 64u + c) * 64u + n];
    Wp[idx] = (__bf16)w;
}

// mid_w (1024,64,64) f32 -> Mp[pos*4096 + (kt*4+nt)*512 + lane*16 + e] bf16
__global__ __launch_bounds__(256) void pack_mid_kernel(
    const float* __restrict__ mid_w, __bf16* __restrict__ Mp)
{
    unsigned idx  = blockIdx.x * 256u + threadIdx.x;   // < 1024*4096
    unsigned e    = idx & 15u;
    unsigned lane = (idx >> 4) & 31u;
    unsigned nt   = (idx >> 9) & 3u;
    unsigned kt   = (idx >> 11) & 1u;
    unsigned pos  = idx >> 12;
    unsigned c    = kt * 32u + ((e >> 3) << 4) + ((lane >> 4) << 3) + (e & 7u);  // 0..63
    unsigned n    = nt * 16u + (lane & 15u);
    float w = mid_w[((size_t)pos * 64u + c) * 64u + n];
    Mp[idx] = (__bf16)w;
}

// ---------------------------------------------------------------------------
// Input projection: V0[b,pos,c] = relu(sum_f x[b,pos,f]*w_in[f,c] + b_in[c])
// ---------------------------------------------------------------------------
__global__ __launch_bounds__(256) void input_proj_kernel(
    const float* __restrict__ in_data, const float* __restrict__ w_in,
    const float* __restrict__ b_in, __bf16* __restrict__ V0)
{
    unsigned idx = blockIdx.x * 256u + threadIdx.x;    // < 512*1024*64
    unsigned c   = idx & 63u;
    unsigned row = idx >> 6;                           // b*1024 + pos
    const float* x = in_data + (size_t)row * IN_F;
    float s = b_in[c];
    #pragma unroll
    for (int f = 0; f < IN_F; ++f) s += x[f] * w_in[f * 64 + c];
    V0[idx] = (__bf16)fmaxf(s, 0.f);
}

// ---------------------------------------------------------------------------
// FUSED two-level butterfly.  grid = (L, K); block = 256 (8 waves).
// WG owns parent node k and a tile (B0..B0+Bt) x (P0..P0+Pt) of grandchild
// positions (Bt*Pt == 128).  Stage 1: children cj=0,1 values at child
// positions [2P0, 2P0+2Pt) -> LDS (relu, bf16, pairs-contiguous).  Stage 2:
// grandchildren gc=0..3 (node 4k+gc) -> global.  LDS: 96KB weights + 64KB
// intermediate = 160KB (dynamic).
// ---------------------------------------------------------------------------
__global__ __launch_bounds__(256) void butterfly_fused2_kernel(
    const __bf16* __restrict__ Vin, __bf16* __restrict__ Vout,
    const __bf16* __restrict__ Wp, const float* __restrict__ biases,
    int L, int lgPt, int lgNpt, int offc, int offg)
{
    extern __shared__ __align__(16) char smemRaw[];
    __bf16* sWc = (__bf16*)smemRaw;            //  2 * 8192  child weights
    __bf16* sWg = sWc + 2 * 8192;              //  4 * 8192  grandchild weights
    __bf16* sV  = sWc + 6 * 8192;              //  2 * 256*64 child values

    const int k   = blockIdx.y;
    const int tx  = blockIdx.x;
    const int Pt  = 1 << lgPt;
    const int Bt  = 128 >> lgPt;
    const int B0  = (tx >> lgNpt) * Bt;
    const int P0  = (tx & ((1 << lgNpt) - 1)) << lgPt;
    const int twoPt = Pt << 1;

    const int t = threadIdx.x;
    // --- async-DMA all 6 weight blocks (child pair + grandchild quad are each
    //     contiguous in Wp); INST_OFFSET applies to LDS and global address ---
    {
        const unsigned lc = lds_offset(sWc) + (unsigned)t * 16u;
        const unsigned lg = lds_offset(sWg) + (unsigned)t * 16u;
        const char* gc_ = (const char*)(Wp + (size_t)(offc + 2 * k) * 8192) + t * 16;
        const char* gg_ = (const char*)(Wp + (size_t)(offg + 4 * k) * 8192) + t * 16;
        asm volatile(
            "global_load_async_to_lds_b128 %0, %2, off\n\t"
            "global_load_async_to_lds_b128 %0, %2, off offset:4096\n\t"
            "global_load_async_to_lds_b128 %0, %2, off offset:8192\n\t"
            "global_load_async_to_lds_b128 %0, %2, off offset:12288\n\t"
            "global_load_async_to_lds_b128 %0, %2, off offset:16384\n\t"
            "global_load_async_to_lds_b128 %0, %2, off offset:20480\n\t"
            "global_load_async_to_lds_b128 %0, %2, off offset:24576\n\t"
            "global_load_async_to_lds_b128 %0, %2, off offset:28672\n\t"
            "global_load_async_to_lds_b128 %1, %3, off\n\t"
            "global_load_async_to_lds_b128 %1, %3, off offset:4096\n\t"
            "global_load_async_to_lds_b128 %1, %3, off offset:8192\n\t"
            "global_load_async_to_lds_b128 %1, %3, off offset:12288\n\t"
            "global_load_async_to_lds_b128 %1, %3, off offset:16384\n\t"
            "global_load_async_to_lds_b128 %1, %3, off offset:20480\n\t"
            "global_load_async_to_lds_b128 %1, %3, off offset:24576\n\t"
            "global_load_async_to_lds_b128 %1, %3, off offset:28672\n\t"
            "global_load_async_to_lds_b128 %1, %3, off offset:32768\n\t"
            "global_load_async_to_lds_b128 %1, %3, off offset:36864\n\t"
            "global_load_async_to_lds_b128 %1, %3, off offset:40960\n\t"
            "global_load_async_to_lds_b128 %1, %3, off offset:45056\n\t"
            "global_load_async_to_lds_b128 %1, %3, off offset:49152\n\t"
            "global_load_async_to_lds_b128 %1, %3, off offset:53248\n\t"
            "global_load_async_to_lds_b128 %1, %3, off offset:57344\n\t"
            "global_load_async_to_lds_b128 %1, %3, off offset:61440\n\t"
            "s_wait_asynccnt 0x0"
            :: "v"(lc), "v"(lg), "v"(gc_), "v"(gg_) : "memory");
    }
    __syncthreads();

    const int wave   = t >> 5;
    const int lane   = t & 31;
    const int mr     = lane & 15;
    const int rowoff = (lane >> 4) << 3;

    // ---- Stage 1: children -> LDS (4 waves per child, 64 rows each) --------
    {
        const int cj = wave >> 2;
        const __bf16* wlds = sWc + cj * 8192;
        __bf16* vdst = sV + cj * (256 * CC);
        const int biasBase = (offc + 2 * k + cj) * CC;
        const int parentBase = k * L + 4 * P0;
        const int lg2Pt = lgPt + 1;
        #pragma unroll
        for (int sub = 0; sub < 2; ++sub) {
            const int s0 = (wave & 3) * 64 + sub * 32;     // row base in [0,256)
            const int sA0 = s0 + mr, sA1 = s0 + 16 + mr;
            const int bl0 = sA0 >> lg2Pt, q0 = sA0 & (twoPt - 1);
            const int bl1 = sA1 >> lg2Pt, q1 = sA1 & (twoPt - 1);
            const __bf16* arow0 =
                Vin + ((size_t)(B0 + bl0) * NPOS + parentBase + 2 * q0) * CC;
            const __bf16* arow1 =
                Vin + ((size_t)(B0 + bl1) * NPOS + parentBase + 2 * q1) * CC;
            v8f acc[2][4];
            zero_acc(acc);
            gemm32x64<4>(arow0, arow1, wlds, lane, acc);
            #pragma unroll
            for (int nt = 0; nt < 4; ++nt) {
                const int n = nt * 16 + mr;
                const float bia = biases[biasBase + n];
                #pragma unroll
                for (int u = 0; u < 2; ++u)
                    #pragma unroll
                    for (int v = 0; v < 8; ++v) {
                        const int row = s0 + u * 16 + rowoff + v;
                        float x = fmaxf(acc[u][nt][v] + bia, 0.f);
                        vdst[row * CC + n] = (__bf16)x;   // LDS, s-major layout
                    }
            }
        }
    }
    __syncthreads();

    // ---- Stage 2: grandchildren -> global (2 waves per gc, 64 rows each) ---
    {
        const int gc = wave >> 1;                   // node 4k+gc, cj = gc>>1
        const __bf16* wlds = sWg + gc * 8192;
        const __bf16* aV = sV + (gc >> 1) * (256 * CC);
        const int biasBase = (offg + 4 * k + gc) * CC;
        const int qL = L >> 2;                      // grandchild node length
        const size_t outBase = (size_t)(4 * k + gc) * qL + P0;
        #pragma unroll
        for (int sub = 0; sub < 2; ++sub) {
            const int r0 = (wave & 1) * 64 + sub * 32;     // row base in [0,128)
            const int rr0 = r0 + mr, rr1 = r0 + 16 + mr;
            const int bl0 = rr0 >> lgPt, p20 = rr0 & (Pt - 1);
            const int bl1 = rr1 >> lgPt, p21 = rr1 & (Pt - 1);
            const __bf16* arow0 = aV + (bl0 * twoPt + 2 * p20) * CC;
            const __bf16* arow1 = aV + (bl1 * twoPt + 2 * p21) * CC;
            v8f acc[2][4];
            zero_acc(acc);
            gemm32x64<4>(arow0, arow1, wlds, lane, acc);
            #pragma unroll
            for (int nt = 0; nt < 4; ++nt) {
                const int n = nt * 16 + mr;
                const float bia = biases[biasBase + n];
                #pragma unroll
                for (int u = 0; u < 2; ++u)
                    #pragma unroll
                    for (int v = 0; v < 8; ++v) {
                        const int r  = r0 + u * 16 + rowoff + v;
                        const int bl = r >> lgPt;
                        const int p2 = r & (Pt - 1);
                        float x = fmaxf(acc[u][nt][v] + bia, 0.f);
                        Vout[((size_t)(B0 + bl) * NPOS + outBase + p2) * CC + n] =
                            (__bf16)x;
                    }
            }
        }
    }
}

// ---------------------------------------------------------------------------
// Single butterfly level (used for L4 and L9).  grid = (L, 2K); 8 waves,
// 32 M-rows per wave; weights async-staged to LDS.
// ---------------------------------------------------------------------------
__global__ __launch_bounds__(256) void butterfly_level_kernel(
    const __bf16* __restrict__ Vin, __bf16* __restrict__ Vout,
    const __bf16* __restrict__ Wp, const float* __restrict__ biases,
    int L, int lgH, int off)
{
    const int child = blockIdx.y;
    const int k     = child >> 1;
    const int j     = child & 1;
    const int cc    = off + child;
    const int halfL = L >> 1;

    __shared__ __align__(16) __bf16 sW[8192];
    {
        const int tt = threadIdx.x;
        const unsigned ldsa = lds_offset(sW) + (unsigned)tt * 16u;
        const char* ga = (const char*)(Wp + (size_t)cc * 8192) + tt * 16;
        asm volatile(
            "global_load_async_to_lds_b128 %0, %1, off\n\t"
            "global_load_async_to_lds_b128 %0, %1, off offset:4096\n\t"
            "global_load_async_to_lds_b128 %0, %1, off offset:8192\n\t"
            "global_load_async_to_lds_b128 %0, %1, off offset:12288\n\t"
            "s_wait_asynccnt 0x0"
            :: "v"(ldsa), "v"(ga) : "memory");
    }
    __syncthreads();

    const int t    = threadIdx.x;
    const int wave = t >> 5;
    const int lane = t & 31;
    const int m0   = blockIdx.x * 256 + wave * 32;
    const int mr   = lane & 15;
    const int r0 = m0 + mr, r1 = m0 + 16 + mr;
    const int b0 = r0 >> lgH, p0 = r0 & (halfL - 1);
    const int b1 = r1 >> lgH, p1 = r1 & (halfL - 1);
    const __bf16* arow0 = Vin + ((size_t)b0 * NPOS + (size_t)k * L + 2 * p0) * CC;
    const __bf16* arow1 = Vin + ((size_t)b1 * NPOS + (size_t)k * L + 2 * p1) * CC;

    v8f acc[2][4];
    zero_acc(acc);
    gemm32x64<4>(arow0, arow1, sW, lane, acc);

    const int rowoff = (lane >> 4) << 3;
    #pragma unroll
    for (int nt = 0; nt < 4; ++nt) {
        const int n     = nt * 16 + mr;
        const float bia = biases[(size_t)cc * CC + n];
        #pragma unroll
        for (int u = 0; u < 2; ++u)
            #pragma unroll
            for (int v = 0; v < 8; ++v) {
                const int r  = m0 + u * 16 + rowoff + v;
                const int b2 = r >> lgH;
                const int p2 = r & (halfL - 1);
                float x = fmaxf(acc[u][nt][v] + bia, 0.f);
                Vout[((size_t)b2 * NPOS + (size_t)k * L + (size_t)j * halfL + p2) * CC + n] =
                    (__bf16)x;
            }
    }
}

// ---------------------------------------------------------------------------
// Mid layer: Out[b,pos,:] = relu(V[b,pos,:] @ mid_w[pos] + mid_b[pos])
// grid = (2, 1024); K-dim = 64 -> 2 WMMA K-steps; 32 M-rows per wave.
// ---------------------------------------------------------------------------
__global__ __launch_bounds__(256) void mid_kernel(
    const __bf16* __restrict__ Vin, __bf16* __restrict__ Vout,
    const __bf16* __restrict__ Mp, const float* __restrict__ mid_b)
{
    const int pos = blockIdx.y;

    __shared__ __align__(16) __bf16 sW[4096];
    {
        const int tt = threadIdx.x;
        const unsigned ldsa = lds_offset(sW) + (unsigned)tt * 16u;
        const char* ga = (const char*)(Mp + (size_t)pos * 4096) + tt * 16;
        asm volatile(
            "global_load_async_to_lds_b128 %0, %1, off\n\t"
            "global_load_async_to_lds_b128 %0, %1, off offset:4096\n\t"
            "s_wait_asynccnt 0x0"
            :: "v"(ldsa), "v"(ga) : "memory");
    }
    __syncthreads();

    const int t    = threadIdx.x;
    const int wave = t >> 5;
    const int lane = t & 31;
    const int m0   = blockIdx.x * 256 + wave * 32;
    const int mr   = lane & 15;
    const int b0 = m0 + mr, b1 = b0 + 16;
    const __bf16* arow0 = Vin + ((size_t)b0 * NPOS + pos) * CC;
    const __bf16* arow1 = Vin + ((size_t)b1 * NPOS + pos) * CC;

    v8f acc[2][4];
    zero_acc(acc);
    gemm32x64<2>(arow0, arow1, sW, lane, acc);

    const int rowoff = (lane >> 4) << 3;
    #pragma unroll
    for (int nt = 0; nt < 4; ++nt) {
        const int n     = nt * 16 + mr;
        const float bia = mid_b[(size_t)pos * CC + n];
        #pragma unroll
        for (int u = 0; u < 2; ++u)
            #pragma unroll
            for (int v = 0; v < 8; ++v) {
                const int b2 = m0 + u * 16 + rowoff + v;
                float x = fmaxf(acc[u][nt][v] + bia, 0.f);
                Vout[((size_t)b2 * NPOS + pos) * CC + n] = (__bf16)x;
            }
    }
}

// ---------------------------------------------------------------------------
// Output projection: out[b, k*8+o] = sum_c V[b,k,c] * w_out[k,c,o]   (fp32)
// ---------------------------------------------------------------------------
__global__ __launch_bounds__(256) void out_proj_kernel(
    const __bf16* __restrict__ V, const float* __restrict__ w_out,
    float* __restrict__ out)
{
    const int kpos = blockIdx.y;
    __shared__ float sw[CC * OUT_F];
    const int t = threadIdx.x;
    sw[t]       = w_out[(size_t)kpos * 512 + t];
    sw[t + 256] = w_out[(size_t)kpos * 512 + 256 + t];
    __syncthreads();

    const int b = blockIdx.x * 256 + t;
    const __bf16* row = V + ((size_t)b * NPOS + kpos) * CC;
    float acc[OUT_F];
    #pragma unroll
    for (int o = 0; o < OUT_F; ++o) acc[o] = 0.f;
    #pragma unroll 8
    for (int c = 0; c < CC; ++c) {
        const float v = (float)row[c];
        #pragma unroll
        for (int o = 0; o < OUT_F; ++o) acc[o] += v * sw[c * OUT_F + o];
    }
    #pragma unroll
    for (int o = 0; o < OUT_F; ++o)
        out[(size_t)b * 8192 + (size_t)kpos * OUT_F + o] = acc[o];
}

// ---------------------------------------------------------------------------
// Host-side orchestration
// ---------------------------------------------------------------------------
extern "C" void kernel_launch(void* const* d_in, const int* in_sizes, int n_in,
                              void* d_out, int out_size, void* d_ws, size_t ws_size,
                              hipStream_t stream)
{
    const float* in_data = (const float*)d_in[0];
    const float* w_in    = (const float*)d_in[1];
    const float* b_in    = (const float*)d_in[2];
    const float* filters = (const float*)d_in[3];
    const float* biases  = (const float*)d_in[4];
    const float* mid_w   = (const float*)d_in[5];
    const float* mid_b   = (const float*)d_in[6];
    const float* w_out   = (const float*)d_in[7];
    float*       out     = (float*)d_out;

    const size_t VBYTES = (size_t)BB * NPOS * CC * 2;            // 67,108,864
    char* ws = (char*)d_ws;
    __bf16* Va = (__bf16*)ws;
    __bf16* Vb = (__bf16*)(ws + VBYTES);
    __bf16* Wp = (__bf16*)(ws + 2 * VBYTES);
    __bf16* Mp = (__bf16*)(ws + 2 * VBYTES + (size_t)TOTAL_BR * 8192 * 2 + (1u << 20));

    const size_t FUSED_LDS = (6 * 8192 + 2 * 256 * CC) * sizeof(__bf16); // 163840

    // 1) pack weights to bf16 WMMA-fragment order
    pack_filters_kernel<<<(TOTAL_BR * 8192) / 256, 256, 0, stream>>>(filters, Wp);
    pack_mid_kernel<<<(NPOS * 4096) / 256, 256, 0, stream>>>(mid_w, Mp);

    // 2) input projection -> Va
    input_proj_kernel<<<(BB * NPOS * CC) / 256, 256, 0, stream>>>(in_data, w_in, b_in, Va);

    // 3) tree: F(0,1) F(2,3) L4 mid F(5,6) F(7,8) L9   (7 activation passes)
    //    level branch offsets: 0,2,6,14,30,62,126,254,510,1022
    butterfly_fused2_kernel<<<dim3(1024, 1), 256, FUSED_LDS, stream>>>(
        Va, Vb, Wp, biases, 1024, /*lgPt*/7, /*lgNpt*/1, /*offc*/0, /*offg*/2);
    butterfly_fused2_kernel<<<dim3(256, 4), 256, FUSED_LDS, stream>>>(
        Vb, Va, Wp, biases, 256, /*lgPt*/6, /*lgNpt*/0, /*offc*/6, /*offg*/14);
    butterfly_level_kernel<<<dim3(64, 32), 256, 0, stream>>>(
        Va, Vb, Wp, biases, /*L*/64, /*lgH*/5, /*off*/30);
    mid_kernel<<<dim3(2, NPOS), 256, 0, stream>>>(Vb, Va, Mp, mid_b);
    butterfly_fused2_kernel<<<dim3(32, 32), 256, FUSED_LDS, stream>>>(
        Va, Vb, Wp, biases, 32, /*lgPt*/3, /*lgNpt*/0, /*offc*/62, /*offg*/126);
    butterfly_fused2_kernel<<<dim3(8, 128), 256, FUSED_LDS, stream>>>(
        Vb, Va, Wp, biases, 8, /*lgPt*/1, /*lgNpt*/0, /*offc*/254, /*offg*/510);
    butterfly_level_kernel<<<dim3(2, 1024), 256, 0, stream>>>(
        Va, Vb, Wp, biases, /*L*/2, /*lgH*/0, /*off*/1022);

    // 4) output projection (fp32)
    out_proj_kernel<<<dim3(2, NPOS), 256, 0, stream>>>(Vb, w_out, out);
}
